// LSTM_75917841924127
// MI455X (gfx1250) — matrix operand order repaired and checked by reading, not compiled, not run on hardware
//
#include <hip/hip_runtime.h>

#define OBS   128
#define HID   1024
#define BATCH 64
#define TSEQ  2048
#define NWG   16
#define OUTCOLS 8192

// Dynamic-LDS layout (bytes). Total 190,464 B < 320 KB WGP LDS.
#define LDS_GW   0          // gate weight B-frags: 8 ks x 16 tiles x 32 lanes x 32B = 131072
#define LDS_PW   131072     // Whr B-frags:         2 ks x  8 tiles x 32 lanes x 32B = 16384
#define LDS_S    147456     // [x_t | h] staging: 64 rows x 264 bf16 (528B rows)    = 33792
#define LDS_PB   181248     // p buffer:          64 rows x  72 bf16 (144B rows)    = 9216
#define LDS_TOTAL 190464

typedef __attribute__((ext_vector_type(16))) __bf16       v16bf;
typedef __attribute__((ext_vector_type(8)))  float        v8f;
typedef __attribute__((ext_vector_type(4)))  unsigned int u32x4;

union FragBF { u32x4 u[2]; v16bf v; };

__device__ __forceinline__ unsigned short f2bf(float f) {
  unsigned u = __builtin_bit_cast(unsigned, f);
  u += 0x7fffu + ((u >> 16) & 1u);            // round-to-nearest-even
  return (unsigned short)(u >> 16);
}
__device__ __forceinline__ float fexp(float x) {
  return __builtin_amdgcn_exp2f(x * 1.4426950408889634f);
}
__device__ __forceinline__ float fsig(float x) {
  return __builtin_amdgcn_rcpf(1.0f + fexp(-x));
}
__device__ __forceinline__ float ftanh(float x) {
  return 1.0f - 2.0f * __builtin_amdgcn_rcpf(1.0f + fexp(2.0f * x));
}

// CDNA5 async global->LDS copy (ASYNCcnt). lds_addr = LDS byte address
// (low 32 bits of the generic pointer: LDS aperture maps addr[31:0]).
// NOTE: the "memory" clobber here (executed every t-iteration for staging)
// is also what stops LICM from hoisting the loop-invariant LDS weight
// fragment loads out of the 2048-step loop (which would spill ~580 VGPRs).
__device__ __forceinline__ void async_g2l(unsigned lds_addr, const void* gaddr) {
  asm volatile("global_load_async_to_lds_b128 %0, %1, off"
               :: "v"(lds_addr), "v"(gaddr) : "memory");
}
__device__ __forceinline__ void wait_async0() {
  asm volatile("s_wait_asynccnt 0" ::: "memory");
}
// plain 16B LDS load -> one ds_load_b128 (keeps addrspace(3) inference).
__device__ __forceinline__ u32x4 lds_ld16(const void* p) {
  return *(const u32x4*)p;
}

struct GBar { unsigned count; unsigned gen; unsigned pad[62]; };

// 16-block sense-reversing device barrier (all 16 WGs co-resident).
__device__ __forceinline__ void grid_barrier(GBar* b) {
  __builtin_amdgcn_fence(__ATOMIC_RELEASE, "agent");
  __syncthreads();
  if (threadIdx.x == 0) {
    unsigned g = __hip_atomic_load(&b->gen, __ATOMIC_RELAXED, __HIP_MEMORY_SCOPE_AGENT);
    unsigned a = __hip_atomic_fetch_add(&b->count, 1u, __ATOMIC_RELAXED, __HIP_MEMORY_SCOPE_AGENT);
    if (a == NWG - 1u) {
      __hip_atomic_store(&b->count, 0u, __ATOMIC_RELAXED, __HIP_MEMORY_SCOPE_AGENT);
      __hip_atomic_fetch_add(&b->gen, 1u, __ATOMIC_RELEASE, __HIP_MEMORY_SCOPE_AGENT);
    } else {
      while (__hip_atomic_load(&b->gen, __ATOMIC_RELAXED, __HIP_MEMORY_SCOPE_AGENT) == g)
        __builtin_amdgcn_s_sleep(1);
    }
  }
  __syncthreads();
  __builtin_amdgcn_fence(__ATOMIC_ACQUIRE, "agent");
}

// ---------------------------------------------------------------------------
// One LSTMP layer, all 2048 steps, 16 blocks x 256 threads (8 waves).
// Block w owns H-slice [w*64, w*64+64) of all 4 gates and the matching
// K-slice of Whr. Weights live in LDS for the whole run; c-state in VGPRs.
// ---------------------------------------------------------------------------
__global__ __launch_bounds__(256)
void lstmp_layer(const unsigned short* __restrict__ xin,   // [B,T,128] bf16
                 const u32x4* __restrict__ gw,             // packed gate B-frags
                 const u32x4* __restrict__ pw,             // packed Whr B-frags
                 const float* __restrict__ bias,           // bih+bhh [4096]
                 float* __restrict__ partials,             // [16][64][128] f32
                 unsigned short* __restrict__ hcur,        // [64][128] bf16
                 unsigned short* __restrict__ h1out,       // bf16 [B,T,128] (layer 0)
                 float* __restrict__ fout,                 // f32 out   (layer 1)
                 GBar* __restrict__ bar,
                 int layer)
{
  extern __shared__ char smem[];

  const int w    = blockIdx.x;
  const int tid  = threadIdx.x;
  const int wv   = tid >> 5;
  const int lane = tid & 31;
  const int half = lane >> 4;
  const int l16  = lane & 15;
  const int m    = wv >> 1;     // M-tile (batch/16)
  const int np   = wv & 1;      // N-pair within 64-wide gate slice

  // ---- one-time: pull this WG's weight fragments into LDS (async DMA) ----
  {
    const u32x4* gwG = gw + (size_t)w * 8192;   // 128 KB slice
    #pragma unroll
    for (int i = 0; i < 32; ++i) {
      const int q = tid + i * 256;
      async_g2l((unsigned)(uintptr_t)(smem + LDS_GW + q * 16), (const void*)(gwG + q));
    }
    const u32x4* pwG = pw + (size_t)w * 1024;   // 16 KB slice
    #pragma unroll
    for (int i = 0; i < 4; ++i) {
      const int q = tid + i * 256;
      async_g2l((unsigned)(uintptr_t)(smem + LDS_PW + q * 16), (const void*)(pwG + q));
    }
    wait_async0();
  }
  __syncthreads();

  // t-invariant bias broadcast values (one scalar per owned output column)
  float biasv[4][2];
  for (int g = 0; g < 4; ++g)
    for (int nn = 0; nn < 2; ++nn)
      biasv[g][nn] = bias[g * HID + w * 64 + (np * 2 + nn) * 16 + l16];

  v8f C[2];                                   // persistent c-state (16 f32/lane)
  for (int nn = 0; nn < 2; ++nn)
    for (int k = 0; k < 8; ++k) C[nn][k] = 0.0f;

  const int srow = tid >> 2;                  // staging row (batch)
  const int sch  = tid & 3;                   // 32-bf16 chunk

  // per-lane LDS base addresses (byte offsets are immediates in the loop)
  const char* gwL = smem + LDS_GW + lane * 32;          // + ((ks*16+tile)*1024)
  const char* pwL = smem + LDS_PW + lane * 32;          // + ((ks*8+wv)*1024)
  const char* aL  = smem + LDS_S + (m * 16 + l16) * 528 + half * 16; // + ks*64 (+32)
  const char* a2L = smem + LDS_PB + l16 * 144 + half * 16;           // + mm*2304 + ks*64 (+32)

  for (int t = 0; t < TSEQ; ++t) {
    // ---- stage x_t (cols 0..127) and h_{t-1} (cols 128..255) via async DMA ----
    {
      const unsigned short* xg = xin + ((size_t)srow * TSEQ + t) * OBS + sch * 32;
      const unsigned short* hg = hcur + srow * OBS + sch * 32;
      const unsigned sb = (unsigned)(uintptr_t)(smem + LDS_S + srow * 528 + sch * 64);
      #pragma unroll
      for (int i = 0; i < 4; ++i) {
        async_g2l(sb + i * 16,       (const void*)(xg + i * 8));
        async_g2l(sb + 256 + i * 16, (const void*)(hg + i * 8));
      }
      wait_async0();
    }
    __syncthreads();

    // ---- gate GEMM: per wave [16 x 256] x [256 x 16] for i,f,g,o x 2 tiles ----
    v8f acc[4][2];
    for (int g = 0; g < 4; ++g)
      for (int nn = 0; nn < 2; ++nn)
        for (int k = 0; k < 8; ++k) acc[g][nn][k] = biasv[g][nn];

    #pragma unroll
    for (int ks = 0; ks < 8; ++ks) {
      FragBF A;
      A.u[0] = lds_ld16(aL + ks * 64);        // K 0..7  / 8..15
      A.u[1] = lds_ld16(aL + ks * 64 + 32);   // K 16..23 / 24..31
      #pragma unroll
      for (int g = 0; g < 4; ++g) {
        #pragma unroll
        for (int nn = 0; nn < 2; ++nn) {
          const int tile = g * 4 + np * 2 + nn;
          const int fb = (ks * 16 + tile) * 1024;   // frag stride = 32 lanes * 32B
          FragBF Bm;
          Bm.u[0] = lds_ld16(gwL + fb);
          Bm.u[1] = lds_ld16(gwL + fb + 16);
          acc[g][nn] = __builtin_amdgcn_wmma_f32_16x16x32_bf16(
              false, A.v, false, Bm.v, (short)0, acc[g][nn], false, false);
        }
      }
    }

    // ---- LSTM cell pointwise; write p = sig(o)*tanh(c) to LDS as bf16 ----
    #pragma unroll
    for (int nn = 0; nn < 2; ++nn) {
      const int colb = (np * 2 + nn) * 16 + l16;
      #pragma unroll
      for (int k = 0; k < 8; ++k) {
        float iv = fsig(acc[0][nn][k]);
        float fv = fsig(acc[1][nn][k]);
        float gv = ftanh(acc[2][nn][k]);
        float ov = fsig(acc[3][nn][k]);
        float c  = fv * C[nn][k] + iv * gv;
        C[nn][k] = c;
        float p  = ov * ftanh(c);
        const int row = m * 16 + k + half * 8;   // C/D tile row layout
        *(unsigned short*)(smem + LDS_PB + row * 144 + colb * 2) = f2bf(p);
      }
    }
    __syncthreads();

    // ---- projection partial: p[64x64] x Whr_sliceT[64x128] -> partials[w] ----
    #pragma unroll
    for (int mm = 0; mm < 4; ++mm) {
      v8f pa = {0.f, 0.f, 0.f, 0.f, 0.f, 0.f, 0.f, 0.f};
      #pragma unroll
      for (int ks = 0; ks < 2; ++ks) {
        FragBF A2;
        A2.u[0] = lds_ld16(a2L + mm * 2304 + ks * 64);
        A2.u[1] = lds_ld16(a2L + mm * 2304 + ks * 64 + 32);
        FragBF Bm;
        Bm.u[0] = lds_ld16(pwL + (ks * 8 + wv) * 1024);
        Bm.u[1] = lds_ld16(pwL + (ks * 8 + wv) * 1024 + 16);
        pa = __builtin_amdgcn_wmma_f32_16x16x32_bf16(
            false, A2.v, false, Bm.v, (short)0, pa, false, false);
      }
      #pragma unroll
      for (int k = 0; k < 8; ++k)
        partials[((size_t)w * 64 + mm * 16 + k + half * 8) * OBS + wv * 16 + l16] = pa[k];
    }

    grid_barrier(bar);   // all partials of step t visible

    // ---- reduce partials over 16 blocks -> h_t (each block does 512 elems) ----
    {
      const int e0 = w * 512 + tid * 2;
      #pragma unroll
      for (int q = 0; q < 2; ++q) {
        const int e = e0 + q;
        float s = 0.0f;
        #pragma unroll
        for (int ww = 0; ww < NWG; ++ww) s += partials[(size_t)ww * 8192 + e];
        hcur[e] = f2bf(s);
        const int b = e >> 7, col = e & 127;
        if (layer == 0) {
          h1out[((size_t)b * TSEQ + t) * OBS + col] = f2bf(s);
        } else if (t >= TSEQ - 64) {
          fout[(size_t)b * OUTCOLS + (t - (TSEQ - 64)) * OBS + col] = s;
        }
      }
    }

    grid_barrier(bar);   // h_t fully published before anyone stages step t+1
  }
}

// ---------------------------------------------------------------------------
// One-time packing / conversion kernels
// ---------------------------------------------------------------------------
// Pack [Wih | Whh] (f32, each [4096,128]) into bf16 WMMA B-fragments:
// dword index d = ((((w*8+ks)*16+tile)*32+lane)*8+j)
__global__ void pack_gatew(const float* __restrict__ Wih, const float* __restrict__ Whh,
                           unsigned* __restrict__ dst, int n) {
  int d = blockIdx.x * blockDim.x + threadIdx.x;
  if (d >= n) return;
  int j = d & 7, lane = (d >> 3) & 31, tile = (d >> 8) & 15, ks = (d >> 12) & 7, w = (d >> 15) & 15;
  int g = tile >> 2, p4 = tile & 3;
  int row = g * HID + w * 64 + p4 * 16 + (lane & 15);       // W row (= B column)
  int K = ks * 32 + 2 * j + 8 * (j >> 2) + 8 * (lane >> 4); // B-matrix K layout
  float lo = (K     < OBS) ? Wih[(size_t)row * OBS + K]     : Whh[(size_t)row * OBS + K - OBS];
  float hi = (K + 1 < OBS) ? Wih[(size_t)row * OBS + K + 1] : Whh[(size_t)row * OBS + K + 1 - OBS];
  dst[d] = (unsigned)f2bf(lo) | ((unsigned)f2bf(hi) << 16);
}

// Pack Whr (f32 [128,1024]) B-fragments: d = (((w*2+ks)*8+ntile)*32+lane)*8+j
__global__ void pack_whr(const float* __restrict__ Whr, unsigned* __restrict__ dst, int n) {
  int d = blockIdx.x * blockDim.x + threadIdx.x;
  if (d >= n) return;
  int j = d & 7, lane = (d >> 3) & 31, ntile = (d >> 8) & 7, ks = (d >> 11) & 1, w = (d >> 12) & 15;
  int ncol = ntile * 16 + (lane & 15);
  int K = w * 64 + ks * 32 + 2 * j + 8 * (j >> 2) + 8 * (lane >> 4);
  float lo = Whr[(size_t)ncol * HID + K];
  float hi = Whr[(size_t)ncol * HID + K + 1];
  dst[d] = (unsigned)f2bf(lo) | ((unsigned)f2bf(hi) << 16);
}

__global__ void add_bias(const float* __restrict__ a, const float* __restrict__ b,
                         float* __restrict__ d, int n) {
  int i = blockIdx.x * blockDim.x + threadIdx.x;
  if (i < n) d[i] = a[i] + b[i];
}

__global__ void conv_bf16(const float* __restrict__ src, unsigned short* __restrict__ dst,
                          size_t n) {
  size_t i = (size_t)blockIdx.x * blockDim.x + threadIdx.x;
  if (i < n) dst[i] = f2bf(src[i]);
}

__global__ void init_state(unsigned* __restrict__ h32, GBar* __restrict__ bar, int n) {
  int i = blockIdx.x * blockDim.x + threadIdx.x;
  if (i < n) h32[i] = 0u;
  if (i == 0) { bar->count = 0u; bar->gen = 0u; }
}

// ---------------------------------------------------------------------------
extern "C" void kernel_launch(void* const* d_in, const int* in_sizes, int n_in,
                              void* d_out, int out_size, void* d_ws, size_t ws_size,
                              hipStream_t stream) {
  (void)in_sizes; (void)n_in; (void)out_size; (void)ws_size;
  const float* x    = (const float*)d_in[0];
  const float* Wih0 = (const float*)d_in[1];
  const float* Whh0 = (const float*)d_in[2];
  const float* bih0 = (const float*)d_in[3];
  const float* bhh0 = (const float*)d_in[4];
  const float* Whr0 = (const float*)d_in[5];
  const float* Wih1 = (const float*)d_in[6];
  const float* Whh1 = (const float*)d_in[7];
  const float* bih1 = (const float*)d_in[8];
  const float* bhh1 = (const float*)d_in[9];
  const float* Whr1 = (const float*)d_in[10];

  char* ws = (char*)d_ws;
  size_t off = 0;
  auto carve = [&](size_t bytes) {
    char* p = ws + off;
    off += (bytes + 255) & ~(size_t)255;
    return p;
  };
  unsigned short* xbf  = (unsigned short*)carve((size_t)BATCH * TSEQ * OBS * 2); // 32 MB
  unsigned short* h1   = (unsigned short*)carve((size_t)BATCH * TSEQ * OBS * 2); // 32 MB
  unsigned*       gw0  = (unsigned*)carve((size_t)524288 * 4);                   //  2 MB
  unsigned*       gw1  = (unsigned*)carve((size_t)524288 * 4);
  unsigned*       pw0  = (unsigned*)carve((size_t)65536 * 4);
  unsigned*       pw1  = (unsigned*)carve((size_t)65536 * 4);
  float*          b0   = (float*)carve((size_t)4096 * 4);
  float*          b1   = (float*)carve((size_t)4096 * 4);
  float*          part = (float*)carve((size_t)NWG * BATCH * OBS * 4);
  unsigned short* hcur = (unsigned short*)carve((size_t)BATCH * OBS * 2);
  GBar*           bar  = (GBar*)carve(sizeof(GBar));

  // allow >64KB dynamic LDS (gfx1250 WGP has 320KB)
  hipFuncSetAttribute((const void*)lstmp_layer,
                      hipFuncAttributeMaxDynamicSharedMemorySize, LDS_TOTAL);

  const size_t nx = (size_t)BATCH * TSEQ * OBS;
  conv_bf16 <<<(unsigned)((nx + 255) / 256), 256, 0, stream>>>(x, xbf, nx);
  pack_gatew<<<524288 / 256, 256, 0, stream>>>(Wih0, Whh0, gw0, 524288);
  pack_gatew<<<524288 / 256, 256, 0, stream>>>(Wih1, Whh1, gw1, 524288);
  pack_whr  <<<65536 / 256, 256, 0, stream>>>(Whr0, pw0, 65536);
  pack_whr  <<<65536 / 256, 256, 0, stream>>>(Whr1, pw1, 65536);
  add_bias  <<<4096 / 256, 256, 0, stream>>>(bih0, bhh0, b0, 4096);
  add_bias  <<<4096 / 256, 256, 0, stream>>>(bih1, bhh1, b1, 4096);
  init_state<<<16, 256, 0, stream>>>((unsigned*)hcur, bar, BATCH * OBS / 2);

  lstmp_layer<<<NWG, 256, LDS_TOTAL, stream>>>(xbf, (const u32x4*)gw0, (const u32x4*)pw0, b0,
                                               part, hcur, h1, (float*)d_out, bar, 0);
  init_state<<<16, 256, 0, stream>>>((unsigned*)hcur, bar, BATCH * OBS / 2);
  lstmp_layer<<<NWG, 256, LDS_TOTAL, stream>>>(h1, (const u32x4*)gw1, (const u32x4*)pw1, b1,
                                               part, hcur, h1, (float*)d_out, bar, 1);
}
